// KnnInvariantPointAttention_7627861918042
// MI455X (gfx1250) — matrix operand order repaired and checked by reading, not compiled, not run on hardware
//
#include <hip/hip_runtime.h>
#include <hip/hip_bf16.h>
#include <stdint.h>

// ---------------- problem constants ----------------
#define NN      8192
#define KK      32
#define C_S     384
#define C_Z     128
#define C_H     16
#define HH      12
#define PQ      4
#define PV      8
#define CZ4     32
#define PROJ_W  1152      // 192 q | 384 kv | 144 qp | 432 kvp
#define ZPROJ_W 48        // 12 b_pair | 32 pair_z | 4 pad
#define CAT_W   960       // 192 o | 96 x | 96 y | 96 z | 96 norm | 384 o_pair
#define INF_V   100000.0f

typedef __attribute__((ext_vector_type(16))) _Float16 v16h;
typedef __attribute__((ext_vector_type(8)))  float    v8f;

// ---------------- WMMA A-fragment loader ----------------
// A: 16x32 f16 fragment from f32 row-major source (inline cvt).
// ISA layout: lanes 0-15 M=lane, lanes16-31 M=lane-16;
// VGPR i: k = (lane>>4)*16 + i*2  (pair k,k+1): contiguous k order chosen here, B packed to match
// The 4 pairs of each group are 8 contiguous floats -> two b128 loads per group.
__device__ inline v16h load_a_f32(const float* __restrict__ A, int lda,
                                  int rowBase, int k0, int lane) {
  int m    = rowBase + (lane & 15);
  int half = lane >> 4;
  const float* base = A + (size_t)m * lda + k0;
  v16h r;
#pragma unroll
  for (int grp = 0; grp < 2; ++grp) {
    const float* b2 = base + half * 16 + grp * 8;   // 16B-aligned
    float4 f0 = *(const float4*)(b2);
    float4 f1 = *(const float4*)(b2 + 4);
    int i0 = grp * 4;
    r[2 * i0 + 0] = (_Float16)f0.x;  r[2 * i0 + 1] = (_Float16)f0.y;
    r[2 * i0 + 2] = (_Float16)f0.z;  r[2 * i0 + 3] = (_Float16)f0.w;
    r[2 * i0 + 4] = (_Float16)f1.x;  r[2 * i0 + 5] = (_Float16)f1.y;
    r[2 * i0 + 6] = (_Float16)f1.z; r[2 * i0 + 7] = (_Float16)f1.w;
  }
  // Pin the per-lane fragment in VGPRs: block any SGPR rematerialization /
  // cross-unroll CSE of the lane-varying A operand (defensive; zero cost).
  asm volatile("" : "+v"(r));
  return r;
}

// ---------------- async staging helpers (CDNA5) ----------------
__device__ inline void async_copy_b128(uint32_t lds_addr, const void* gaddr) {
  asm volatile("global_load_async_to_lds_b128 %0, %1, off"
               :: "v"(lds_addr), "v"((uint64_t)(uintptr_t)gaddr)
               : "memory");
}
__device__ inline void wait_async0() {
  asm volatile("s_wait_asynccnt 0x0" ::: "memory");
}

// ---------------- generic WMMA GEMM:  C = A(f32) * Bpk(f16, fragment-swizzled) + bias ---
// Packed-B layout (built by pack_w): tile (kchunk, c16) = 256 dwords at
//   ((kchunk*ntiles + c16) * 256) ordered [n(16)][half(2)][i(8)], so each lane's
//   B fragment is 32 contiguous bytes -> 2 x ds_load_b128, no shuffles.
// Block = 256 threads = 8 waves; each wave owns MT=2 M-tiles x NT column tiles
// (32 rows x NT*16 cols) -> 2*NT WMMAs per k-step. The NT-KB B chunk is
// async-copied to LDS once per block per k-step (double buffered).
template <int NT>
__global__ __launch_bounds__(256) void gemm_wmma(
    const float* __restrict__ A, int lda,
    const uint32_t* __restrict__ Bpk, int ntiles,
    float* __restrict__ C, int ldc,
    const float* __restrict__ bias,
    int Kd, int strips) {
  constexpr int MT    = 2;                  // M tiles per wave
  constexpr int CHUNK = NT * 256;           // dwords per k-step chunk
  constexpr int XFERS = CHUNK / 4;          // 16B transfers per chunk
  __shared__ __align__(16) uint32_t ldsB[2][CHUNK];

  int blockMtile = blockIdx.x / strips;     // in units of 8 waves * MT tiles
  int strip      = blockIdx.x % strips;
  int w    = threadIdx.x >> 5;
  int lane = threadIdx.x & 31;
  int rowBase = (blockMtile * 8 + w) * (16 * MT);

  // stage the B chunk for k-step k0 into ldsB[buf] (linear 16B bursts)
  auto stage = [&](int k0, int buf) {
    int t = threadIdx.x;
    if (t < XFERS) {
      const uint32_t* g = Bpk + (((size_t)(k0 >> 5) * ntiles + strip * NT) << 8) + t * 4;
      async_copy_b128((uint32_t)(uintptr_t)&ldsB[buf][t * 4], g);
    }
  };

  stage(0, 0);
  wait_async0();
  __syncthreads();

  int n_lane = lane & 15, half = lane >> 4;
  v8f acc[MT][NT] = {};
  int buf = 0;
  for (int k0 = 0; k0 < Kd; k0 += 32) {
    bool more = (k0 + 32) < Kd;
    if (more) stage(k0 + 32, buf ^ 1);         // overlaps with compute below

    if (more) {  // prefetch next A chunk for this lane's rows
#pragma unroll
      for (int mt = 0; mt < MT; ++mt) {
        const float* pf =
            A + (size_t)(rowBase + mt * 16 + n_lane) * lda + k0 + 32 + half * 8;
        __builtin_prefetch(pf, 0, 3);
      }
    }
    v16h a[MT];
#pragma unroll
    for (int mt = 0; mt < MT; ++mt)
      a[mt] = load_a_f32(A, lda, rowBase + mt * 16, k0, lane);
#pragma unroll
    for (int tt = 0; tt < NT; ++tt) {
      union { v16h v; uint4 q[2]; } bf;
      const uint4* p = (const uint4*)&ldsB[buf][tt * 256 + n_lane * 16 + half * 8];
      bf.q[0] = p[0];
      bf.q[1] = p[1];
#pragma unroll
      for (int mt = 0; mt < MT; ++mt)
        acc[mt][tt] = __builtin_amdgcn_wmma_f32_16x16x32_f16(
            false, a[mt], false, bf.v, (short)0, acc[mt][tt], false, false);
    }

    wait_async0();        // next chunk landed (overlapped with the WMMAs above)
    __syncthreads();      // all waves done reading buf before it is overwritten
    buf ^= 1;
  }

  // D layout: lane&15 = n, VGPR r -> M = r + 8*(lane>>4)
  int colBase0 = strip * (NT * 16);
#pragma unroll
  for (int mt = 0; mt < MT; ++mt) {
#pragma unroll
    for (int tt = 0; tt < NT; ++tt) {
      int col = colBase0 + tt * 16 + n_lane;
      float bv = bias[col];
#pragma unroll
      for (int r = 0; r < 8; ++r) {
        int row = rowBase + mt * 16 + r + 8 * half;
        C[(size_t)row * ldc + col] = acc[mt][tt][r] + bv;
      }
    }
  }
}

// ---------------- weight pack: f16 pair-pack + fragment swizzle ----------------
// dst dword offset for (pair-row kp, global col cg), Nd total cols:
//   kchunk=kp/16, rowL=kp%16 -> half=rowL/8, i=rowL%8 ; c16=cg/16, n=cg%16
//   off = ((kchunk*(Nd/16) + c16)*16 + n)*16 + half*8 + i
__global__ void pack_w(const float* __restrict__ src, uint32_t* __restrict__ dst,
                       int Ksrc, int Nsrc, int Nd, int colOff) {
  int idx = blockIdx.x * blockDim.x + threadIdx.x;
  int total = (Ksrc >> 1) * Nsrc;
  if (idx >= total) return;
  int kp = idx / Nsrc, col = idx % Nsrc;
  union { _Float16 h[2]; uint32_t u; } p;
  p.h[0] = (_Float16)src[(size_t)(2 * kp) * Nsrc + col];
  p.h[1] = (_Float16)src[(size_t)(2 * kp + 1) * Nsrc + col];
  int cg = colOff + col;
  int kchunk = kp >> 4, rowL = kp & 15;
  int half = rowL >> 3, i = rowL & 7;
  int c16 = cg >> 4, n = cg & 15;
  size_t off = ((size_t)(kchunk * (Nd >> 4) + c16) * 16 + n) * 16 + half * 8 + i;
  dst[off] = p.u;
}

__global__ void copy_f32(const float* __restrict__ src, float* __restrict__ dst,
                         int n, int off) {
  int i = blockIdx.x * blockDim.x + threadIdx.x;
  if (i < n) dst[off + i] = src[i];
}

__global__ void zero_f32(float* __restrict__ dst, int n) {
  int i = blockIdx.x * blockDim.x + threadIdx.x;
  if (i < n) dst[i] = 0.0f;
}

// ---------------- rigid transform: proj point cols -> global-frame points ----------------
__global__ void rigid_fwd(const float* __restrict__ proj,
                          const float* __restrict__ rot,
                          const float* __restrict__ trans,
                          float* __restrict__ qpts,
                          float* __restrict__ kvpts) {
  int idx = blockIdx.x * blockDim.x + threadIdx.x;   // NN * 192
  int n = idx / 192, p = idx % 192;
  if (n >= NN) return;
  const float* R = rot + (size_t)n * 9;
  const float* T = trans + (size_t)n * 3;
  float x0, x1, x2;
  if (p < 48) {
    const float* row = proj + (size_t)n * PROJ_W + 576;
    x0 = row[p]; x1 = row[48 + p]; x2 = row[96 + p];
  } else {
    int pp = p - 48;
    const float* row = proj + (size_t)n * PROJ_W + 720;
    x0 = row[pp]; x1 = row[144 + pp]; x2 = row[288 + pp];
  }
  float o0 = R[0] * x0 + R[1] * x1 + R[2] * x2 + T[0];
  float o1 = R[3] * x0 + R[4] * x1 + R[5] * x2 + T[1];
  float o2 = R[6] * x0 + R[7] * x1 + R[8] * x2 + T[2];
  float* d = (p < 48) ? qpts + ((size_t)n * 48 + p) * 3
                      : kvpts + ((size_t)n * 144 + (p - 48)) * 3;
  d[0] = o0; d[1] = o1; d[2] = o2;
}

// ---------------- fused attention core ----------------
// grid = NN blocks, block = 384 threads = 12 waves; wave h = head, lane k = neighbor.
__global__ __launch_bounds__(384) void attn_fused(
    const float* __restrict__ proj,   // [NN][1152]
    const float* __restrict__ zproj,  // [NN*KK][48]
    const float* __restrict__ qpts,   // [NN][48][3]
    const float* __restrict__ kvpts,  // [NN][144][3]
    const int*   __restrict__ edge_index,  // [NN][KK]
    const float* __restrict__ rot,    // [NN][3][3]
    const float* __restrict__ trans,  // [NN][3]
    const float* __restrict__ mask,   // [NN]
    const float* __restrict__ head_weights, // [HH]
    float* __restrict__ cat) {        // [NN][960]
  __shared__ float sA[HH][KK];            // softmax probs
  __shared__ float sV[KK][HH][C_H];       // gathered v
  __shared__ float sVP[KK][HH][PV * 3];   // gathered v_pts (global frame)
  __shared__ int   sE[KK];
  __shared__ float sEM[KK];
  __shared__ float sR[9];
  __shared__ float sT[3];

  int n = blockIdx.x;
  int t = threadIdx.x;
  int h = t >> 5;
  int k = t & 31;

  if (t < 9)  sR[t] = rot[(size_t)n * 9 + t];
  if (t >= 9 && t < 12) sT[t - 9] = trans[(size_t)n * 3 + (t - 9)];
  if (h == 0) {
    int e = edge_index[(size_t)n * KK + k];
    sE[k]  = e;
    sEM[k] = mask[e];
  }
  __syncthreads();

  int e = sE[k];
  // q . k_g and stage v
  const float* qrow  = proj + (size_t)n * PROJ_W + h * C_H;
  const float* kvrow = proj + (size_t)e * PROJ_W + 192 + h * (2 * C_H);
  float qk = 0.f;
#pragma unroll
  for (int c = 0; c < C_H; ++c) {
    qk += qrow[c] * kvrow[c];
    sV[k][h][c] = kvrow[C_H + c];
  }
  // point distance term and stage v_pts
  const float* qp = qpts + ((size_t)n * 48 + h * PQ) * 3;
  const float* kp = kvpts + ((size_t)e * 144 + h * (PQ + PV)) * 3;
  float ptsum = 0.f;
#pragma unroll
  for (int pq = 0; pq < PQ; ++pq) {
    float dx = qp[pq * 3 + 0] - kp[pq * 3 + 0];
    float dy = qp[pq * 3 + 1] - kp[pq * 3 + 1];
    float dz = qp[pq * 3 + 2] - kp[pq * 3 + 2];
    ptsum += dx * dx + dy * dy + dz * dz;
  }
  const float* vp = kvpts + ((size_t)e * 144 + h * (PQ + PV) + PQ) * 3;
#pragma unroll
  for (int i = 0; i < PV * 3; ++i) sVP[k][h][i] = vp[i];

  // pair bias
  const float* zr = zproj + (size_t)(n * KK + k) * ZPROJ_W;
  float bpair = zr[h];

  // head weight: softplus(hw) * sqrt(1/(3*(PQ*9/2)))
  float hw  = head_weights[h];
  float sp  = log1pf(expf(hw));
  float wpt = sp * sqrtf(1.0f / 54.0f);

  float logit = qk * sqrtf(1.0f / (3.0f * C_H))
              + sqrtf(1.0f / 3.0f) * bpair
              - 0.5f * wpt * ptsum
              + INF_V * (sEM[k] - 1.0f);

  // wave32 softmax over neighbors (lanes)
  float m = logit;
#pragma unroll
  for (int off = 16; off; off >>= 1) m = fmaxf(m, __shfl_xor(m, off, 32));
  float ex = expf(logit - m);
  float ssum = ex;
#pragma unroll
  for (int off = 16; off; off >>= 1) ssum += __shfl_xor(ssum, off, 32);
  float a = ex / ssum;
  sA[h][k] = a;
  __syncthreads();

  float* cat_n = cat + (size_t)n * CAT_W;
  if (t < 96) {
    // point outputs: (h2, p) -> inverse rigid + norm
    int h2 = t >> 3, p = t & 7;
    float g0 = 0.f, g1 = 0.f, g2 = 0.f;
#pragma unroll 4
    for (int kk = 0; kk < KK; ++kk) {
      float av = sA[h2][kk];
      const float* v3 = &sVP[kk][h2][p * 3];
      g0 += av * v3[0]; g1 += av * v3[1]; g2 += av * v3[2];
    }
    g0 -= sT[0]; g1 -= sT[1]; g2 -= sT[2];
    float l0 = sR[0] * g0 + sR[3] * g1 + sR[6] * g2;   // R^T
    float l1 = sR[1] * g0 + sR[4] * g1 + sR[7] * g2;
    float l2 = sR[2] * g0 + sR[5] * g1 + sR[8] * g2;
    float nm = sqrtf(l0 * l0 + l1 * l1 + l2 * l2 + 1e-8f);
    int hp = h2 * PV + p;
    cat_n[192 + hp] = l0;
    cat_n[288 + hp] = l1;
    cat_n[384 + hp] = l2;
    cat_n[480 + hp] = nm;
  } else if (t < 288) {
    // o[h2][c]
    int o = t - 96;
    int h2 = o >> 4, c = o & 15;
    float acc = 0.f;
#pragma unroll 4
    for (int kk = 0; kk < KK; ++kk) acc += sA[h2][kk] * sV[kk][h2][c];
    cat_n[h2 * C_H + c] = acc;
  } else {
    // o_pair: 384 elems over 96 threads, pair_z read from zproj (n-local, cached)
    int t0 = t - 288;
    for (int j = t0; j < HH * CZ4; j += 96) {
      int h2 = j >> 5, c = j & 31;
      float acc = 0.f;
      for (int kk = 0; kk < KK; ++kk)
        acc += sA[h2][kk] * zproj[(size_t)(n * KK + kk) * ZPROJ_W + 12 + c];
      cat_n[576 + j] = acc;
    }
  }
}

// ---------------- launch ----------------
static inline int cdiv(int a, int b) { return (a + b - 1) / b; }

extern "C" void kernel_launch(void* const* d_in, const int* in_sizes, int n_in,
                              void* d_out, int out_size, void* d_ws, size_t ws_size,
                              hipStream_t stream) {
  const float* s          = (const float*)d_in[0];
  const float* z          = (const float*)d_in[1];
  const int*   edge_index = (const int*)d_in[2];
  const float* rot        = (const float*)d_in[3];
  const float* trans      = (const float*)d_in[4];
  const float* mask       = (const float*)d_in[5];
  const float* w_q  = (const float*)d_in[6];
  const float* b_q  = (const float*)d_in[7];
  const float* w_kv = (const float*)d_in[8];
  const float* b_kv = (const float*)d_in[9];
  const float* w_qp = (const float*)d_in[10];
  const float* b_qp = (const float*)d_in[11];
  const float* w_kvp = (const float*)d_in[12];
  const float* b_kvp = (const float*)d_in[13];
  const float* w_b  = (const float*)d_in[14];
  const float* b_b  = (const float*)d_in[15];
  const float* w_dz = (const float*)d_in[16];
  const float* b_dz = (const float*)d_in[17];
  const float* head_weights = (const float*)d_in[18];
  const float* w_out = (const float*)d_in[19];
  const float* b_out = (const float*)d_in[20];
  float* out = (float*)d_out;

  // workspace carve-up (all 256B aligned); total ~140.1 MB
  char* ws = (char*)d_ws;
  size_t off = 0;
  auto take = [&](size_t bytes) {
    void* p = ws + off;
    off += (bytes + 255) & ~(size_t)255;
    return p;
  };
  float*    proj    = (float*)take((size_t)NN * PROJ_W * 4);        // 37.7 MB
  float*    zproj   = (float*)take((size_t)NN * KK * ZPROJ_W * 4);  // 50.3 MB
  float*    qpts    = (float*)take((size_t)NN * 48 * 3 * 4);        //  4.7 MB
  float*    kvpts   = (float*)take((size_t)NN * 144 * 3 * 4);       // 14.2 MB
  float*    cat     = (float*)take((size_t)NN * CAT_W * 4);         // 31.5 MB
  uint32_t* wcat_pk = (uint32_t*)take((size_t)(C_S / 2) * PROJ_W * 4);
  float*    bcat    = (float*)take(PROJ_W * 4);
  uint32_t* wz_pk   = (uint32_t*)take((size_t)(C_Z / 2) * ZPROJ_W * 4);
  float*    bz      = (float*)take(ZPROJ_W * 4);
  uint32_t* wout_pk = (uint32_t*)take((size_t)(CAT_W / 2) * C_S * 4);

  // --- pack weights (pair-packed + fragment-swizzled f16) + concat biases ---
  zero_f32<<<cdiv((C_Z / 2) * ZPROJ_W, 256), 256, 0, stream>>>((float*)wz_pk, (C_Z / 2) * ZPROJ_W);
  zero_f32<<<1, 256, 0, stream>>>(bz, ZPROJ_W);

  pack_w<<<cdiv(192 * 192, 256), 256, 0, stream>>>(w_q,   wcat_pk, C_S, 192, PROJ_W, 0);
  pack_w<<<cdiv(192 * 384, 256), 256, 0, stream>>>(w_kv,  wcat_pk, C_S, 384, PROJ_W, 192);
  pack_w<<<cdiv(192 * 144, 256), 256, 0, stream>>>(w_qp,  wcat_pk, C_S, 144, PROJ_W, 576);
  pack_w<<<cdiv(192 * 432, 256), 256, 0, stream>>>(w_kvp, wcat_pk, C_S, 432, PROJ_W, 720);
  pack_w<<<cdiv(64 * 12, 256),   256, 0, stream>>>(w_b,   wz_pk,   C_Z, 12,  ZPROJ_W, 0);
  pack_w<<<cdiv(64 * 32, 256),   256, 0, stream>>>(w_dz,  wz_pk,   C_Z, 32,  ZPROJ_W, 12);
  pack_w<<<cdiv(480 * 384, 256), 256, 0, stream>>>(w_out, wout_pk, CAT_W, C_S, C_S, 0);

  copy_f32<<<1, 256, 0, stream>>>(b_q,  bcat, 192, 0);
  copy_f32<<<2, 256, 0, stream>>>(b_kv, bcat, 384, 192);
  copy_f32<<<1, 256, 0, stream>>>(b_qp, bcat, 144, 576);
  copy_f32<<<2, 256, 0, stream>>>(b_kvp, bcat, 432, 720);
  copy_f32<<<1, 64, 0, stream>>>(b_b,  bz, 12, 0);
  copy_f32<<<1, 64, 0, stream>>>(b_dz, bz, 32, 12);

  // --- GEMM P: proj = s @ wcat + bcat   (8192 x 384 x 1152) ---
  gemm_wmma<4><<<32 * 18, 256, 0, stream>>>(s, C_S, wcat_pk, PROJ_W / 16, proj, PROJ_W, bcat, C_S, PROJ_W / 64);

  // --- GEMM Z: zproj = z @ wz + bz      (262144 x 128 x 48) ---
  gemm_wmma<3><<<1024, 256, 0, stream>>>(z, C_Z, wz_pk, ZPROJ_W / 16, zproj, ZPROJ_W, bz, C_Z, 1);

  // --- rigid transform ---
  rigid_fwd<<<(NN * 192) / 256, 256, 0, stream>>>(proj, rot, trans, qpts, kvpts);

  // --- fused attention ---
  attn_fused<<<NN, 384, 0, stream>>>(proj, zproj, qpts, kvpts, edge_index,
                                     rot, trans, mask, head_weights, cat);

  // --- GEMM O: out = cat @ w_out + b_out (8192 x 960 x 384) ---
  gemm_wmma<4><<<32 * 6, 256, 0, stream>>>(cat, CAT_W, wout_pk, C_S / 16, out, C_S, b_out, CAT_W, C_S / 64);
}
